// RelMultiHeadAttn_86028194939537
// MI455X (gfx1250) — matrix operand order repaired and checked by reading, not compiled, not run on hardware
//
#include <hip/hip_runtime.h>
#include <math.h>

#define S   1024
#define BB  8
#define NH  8
#define DH  32
#define HH  256   // NH*DH

typedef __attribute__((ext_vector_type(16))) __bf16 v16bf;
typedef __attribute__((ext_vector_type(8)))  float  v8f;

__device__ __forceinline__ v8f wmma_bf16(v16bf a, v16bf b, v8f c) {
    // (neg_a, A, neg_b, B, c_mod, C, reuse_a, reuse_b)
    return __builtin_amdgcn_wmma_f32_16x16x32_bf16(false, a, false, b, (short)0, c, false, false);
}

__device__ __forceinline__ float wave_max(float v) {
#pragma unroll
    for (int off = 16; off > 0; off >>= 1)
        v = fmaxf(v, __shfl_xor(v, off, 32));
    return v;
}
__device__ __forceinline__ float wave_sum(float v) {
#pragma unroll
    for (int off = 16; off > 0; off >>= 1)
        v += __shfl_xor(v, off, 32);
    return v;
}

// A-fragment (16-bit, 16x32, M=query rows, K=d): contiguous K order used here (B loads match) —
// lanes 0-15: M=lane, K {0..15}; lanes 16-31: M=lane-16, K {16..31}
__device__ __forceinline__ v16bf load_a_q(const float* __restrict__ q,
                                          const float* __restrict__ bias,
                                          int ibase, int b, int n, int lane) {
    const int hi  = (lane >= 16) ? 1 : 0;
    const int row = lane & 15;
    int gi = ibase + row; if (gi > S - 1) gi = S - 1;   // clamped; extra row unused at edge
    const float* qr = q + ((size_t)gi * BB + b) * HH + n * DH;
    const float* br = bias + n * DH;
    v16bf a;
#pragma unroll
    for (int p = 0; p < 8; ++p) {
        const int kb = (hi ? 16 : 0) + 2 * p;
        a[2 * p]     = (__bf16)(qr[kb]     + br[kb]);
        a[2 * p + 1] = (__bf16)(qr[kb + 1] + br[kb + 1]);
    }
    return a;
}

// B-fragment (16-bit, 32x16, K=d, N=key/rel column) from K tensor
__device__ __forceinline__ v16bf load_b_k(const float* __restrict__ k,
                                          int ct, int b, int n, int lane) {
    const int hi = (lane >= 16) ? 1 : 0;
    const int j  = ct * 16 + (lane & 15);
    const float* kr = k + ((size_t)j * BB + b) * HH + n * DH + (hi ? 16 : 0);
    v16bf bf;
#pragma unroll
    for (int p = 0; p < 8; ++p) {
        bf[2 * p]     = (__bf16)kr[2 * p];
        bf[2 * p + 1] = (__bf16)kr[2 * p + 1];
    }
    return bf;
}

// B-fragment from R tensor (r: [S,1,HH])
__device__ __forceinline__ v16bf load_b_r(const float* __restrict__ r,
                                          int ct, int n, int lane) {
    const int hi = (lane >= 16) ? 1 : 0;
    const int j  = ct * 16 + (lane & 15);
    const float* rr = r + (size_t)j * HH + n * DH + (hi ? 16 : 0);
    v16bf bf;
#pragma unroll
    for (int p = 0; p < 8; ++p) {
        bf[2 * p]     = (__bf16)rr[2 * p];
        bf[2 * p + 1] = (__bf16)rr[2 * p + 1];
    }
    return bf;
}

// B-fragment (K=j within 32-chunk, N=d column) from LDS-staged V_n (bf16 [S][DH])
__device__ __forceinline__ v16bf load_b_v_lds(const __bf16* __restrict__ vbuf,
                                              int kc, int dt, int lane) {
    const int hi = (lane >= 16) ? 1 : 0;
    const int d  = dt * 16 + (lane & 15);
    const int jb = kc * 32 + (hi ? 16 : 0);
    v16bf bf;
#pragma unroll
    for (int p = 0; p < 8; ++p) {
        const int j0 = jb + 2 * p;
        bf[2 * p]     = vbuf[j0 * DH + d];
        bf[2 * p + 1] = vbuf[(j0 + 1) * DH + d];
    }
    return bf;
}

// A-fragment of softmax weights: exp-normalize score rows on the fly
__device__ __forceinline__ v16bf load_a_w(const float* __restrict__ sc,
                                          const float* __restrict__ rowm,
                                          const float* __restrict__ rowinv,
                                          int kc, int lane) {
    const int hi  = (lane >= 16) ? 1 : 0;
    const int row = lane & 15;
    const float m = rowm[row], inv = rowinv[row];
    v16bf a;
#pragma unroll
    for (int p = 0; p < 8; ++p) {
        const int kb = (hi ? 16 : 0) + 2 * p;
        const int j  = kc * 32 + kb;
        a[2 * p]     = (__bf16)(__expf(sc[row * S + j]     - m) * inv);
        a[2 * p + 1] = (__bf16)(__expf(sc[row * S + j + 1] - m) * inv);
    }
    return a;
}

__global__ void __launch_bounds__(256)
relattn_kernel(const float* __restrict__ q, const float* __restrict__ k,
               const float* __restrict__ v, const float* __restrict__ r,
               const float* __restrict__ bias_u, const float* __restrict__ bias_v,
               const unsigned char* __restrict__ mask,
               float* __restrict__ out, float* __restrict__ probs) {
    extern __shared__ float smem[];
    float*  sc       = smem;                    // 16*S  : scores for current head
    float*  logit    = sc + 16 * S;             // 16*S  : sum over heads (for probs)
    float*  G        = logit + 16 * S;          // 17*S  : BD_raw rows i0..i0+16
    __bf16* vbuf     = (__bf16*)G;              // S*DH bf16 (64KB) aliased over G (68KB)
    float*  partials = G + 17 * S;              // 8*16*32
    float*  rowm     = partials + 8 * 16 * 32;  // 16
    float*  rowinv   = rowm + 16;               // 16

    const int i0   = blockIdx.x * 16;
    const int b    = blockIdx.y;
    const int tid  = threadIdx.x;
    const int w    = tid >> 5;
    const int lane = tid & 31;
    const int hi   = (lane >= 16) ? 1 : 0;
    const float SCALE = 0.17677669529663687f;  // 1/sqrt(32)
    const float NEGINF = -__builtin_inff();

    for (int n = 0; n < NH; ++n) {
        v16bf a_u  = load_a_q(q, bias_u, i0,     b, n, lane);
        v16bf a_v0 = load_a_q(q, bias_v, i0,     b, n, lane);
        v16bf a_v1 = load_a_q(q, bias_v, i0 + 1, b, n, lane);

        // ---- Phase A: G = (q+bias_v) . R^T for rows i0..i0+16 ----
        for (int ct = w; ct < S / 16; ct += 8) {
            v16bf br = load_b_r(r, ct, n, lane);
            v8f c0 = {}; c0 = wmma_bf16(a_v0, br, c0);
            v8f c1 = {}; c1 = wmma_bf16(a_v1, br, c1);
            const int col = ct * 16 + (lane & 15);
#pragma unroll
            for (int m = 0; m < 8; ++m)
                G[(m + hi * 8) * S + col] = c0[m];
#pragma unroll
            for (int m = 0; m < 8; ++m)
                G[(m + hi * 8 + 1) * S + col] = c1[m];
        }
        __syncthreads();

        // ---- Phase B: AC via WMMA + rel-shifted BD gather + tanh clip ----
        for (int ct = w; ct < S / 16; ct += 8) {
            if (ct + 8 < S / 16) {  // prefetch next K tile (global_prefetch_b8)
                const int jn = (ct + 8) * 16 + (lane & 15);
                __builtin_prefetch(k + ((size_t)jn * BB + b) * HH + n * DH + (hi ? 16 : 0), 0, 1);
            }
            v16bf bk = load_b_k(k, ct, b, n, lane);
            v8f c = {};
            c = wmma_bf16(a_u, bk, c);
            const int j = ct * 16 + (lane & 15);
#pragma unroll
            for (int m = 0; m < 8; ++m) {
                const int il = m + hi * 8;
                const int i  = i0 + il;
                float bd;
                if (j <= i)          bd = G[il * S + (S - 1 - i + j)];
                else if (j == i + 1) bd = 0.0f;
                else                 bd = G[(il + 1) * S + (j - i - 2)];
                float s = (c[m] + bd) * SCALE;
                s = 10.0f * tanhf(s);
                sc[il * S + j] = s;
                if (n == 0) logit[il * S + j] = s;
                else        logit[il * S + j] += s;
            }
        }
        __syncthreads();   // G dead from here; vbuf may overwrite it

        // ---- Phase C: softmax stats (rows 2w,2w+1) + coalesced V_n staging ----
        for (int rsel = 0; rsel < 2; ++rsel) {
            const int row = 2 * w + rsel;
            float m = NEGINF;
            for (int j = lane; j < S; j += 32) m = fmaxf(m, sc[row * S + j]);
            m = wave_max(m);
            float ss = 0.f;
            for (int j = lane; j < S; j += 32) ss += __expf(sc[row * S + j] - m);
            ss = wave_sum(ss);
            if (lane == 0) { rowm[row] = m; rowinv[row] = 1.0f / ss; }
        }
        // stage V_n -> LDS bf16 with coalesced float4 loads
        for (int idx = tid; idx < S * DH / 4; idx += 256) {
            const int j  = idx >> 3;        // row (8 float4 per 32-wide row)
            const int c4 = idx & 7;
            const float4 val = *(const float4*)&v[((size_t)j * BB + b) * HH + n * DH + c4 * 4];
            __bf16* dst = vbuf + j * DH + c4 * 4;
            dst[0] = (__bf16)val.x; dst[1] = (__bf16)val.y;
            dst[2] = (__bf16)val.z; dst[3] = (__bf16)val.w;
        }
        __syncthreads();

        // ---- Phase D: out = softmax(score) . V via WMMA, split over k-chunks ----
        v8f o0 = {}, o1 = {};
        for (int kc = w; kc < S / 32; kc += 8) {
            v16bf aw = load_a_w(sc, rowm, rowinv, kc, lane);
            v16bf b0 = load_b_v_lds(vbuf, kc, 0, lane);
            v16bf b1 = load_b_v_lds(vbuf, kc, 1, lane);
            o0 = wmma_bf16(aw, b0, o0);
            o1 = wmma_bf16(aw, b1, o1);
        }
        {
            const int col = lane & 15;
#pragma unroll
            for (int m = 0; m < 8; ++m) {
                const int il = m + hi * 8;
                partials[(w * 16 + il) * 32 + col]      = o0[m];
                partials[(w * 16 + il) * 32 + 16 + col] = o1[m];
            }
        }
        __syncthreads();
        for (int idx = tid; idx < 16 * 32; idx += 256) {
            const int il = idx >> 5, d = idx & 31;
            float acc = 0.f;
#pragma unroll
            for (int ww = 0; ww < 8; ++ww) acc += partials[(ww * 16 + il) * 32 + d];
            out[((size_t)(i0 + il) * BB + b) * HH + n * DH + d] = acc;
        }
        __syncthreads();
    }

    // ---- probs: masked softmax of head-mean logits ----
    for (int rsel = 0; rsel < 2; ++rsel) {
        const int row = 2 * w + rsel;
        const int i   = i0 + row;
        float m = NEGINF;
        for (int j = lane; j < S; j += 32) {
            float lv = logit[row * S + j] * 0.125f;
            const unsigned char mk = mask[((size_t)i * S + j) * BB + b];
            const float val = mk ? NEGINF : lv;
            logit[row * S + j] = val;
            m = fmaxf(m, val);
        }
        m = wave_max(m);
        float ss = 0.f;
        for (int j = lane; j < S; j += 32) ss += __expf(logit[row * S + j] - m);
        ss = wave_sum(ss);
        const float inv = 1.0f / ss;
        for (int j = lane; j < S; j += 32)
            probs[((size_t)i * S + j) * BB + b] = __expf(logit[row * S + j] - m) * inv;
    }
}

extern "C" void kernel_launch(void* const* d_in, const int* in_sizes, int n_in,
                              void* d_out, int out_size, void* d_ws, size_t ws_size,
                              hipStream_t stream) {
    (void)in_sizes; (void)n_in; (void)out_size; (void)d_ws; (void)ws_size;
    const float*         q   = (const float*)d_in[0];
    const float*         k   = (const float*)d_in[1];
    const float*         v   = (const float*)d_in[2];
    const float*         r   = (const float*)d_in[3];
    const float*         bu  = (const float*)d_in[4];
    const float*         bv  = (const float*)d_in[5];
    const unsigned char* msk = (const unsigned char*)d_in[6];

    float* out   = (float*)d_out;                       // [S, B, HH]
    float* probs = out + (size_t)S * BB * HH;           // [S, S, B]

    const size_t shmem = (size_t)(16 * S + 16 * S + 17 * S + 8 * 16 * 32 + 32) * sizeof(float);
    dim3 grid(S / 16, BB);
    relattn_kernel<<<grid, 256, shmem, stream>>>(q, k, v, r, bu, bv, msk, out, probs);
}
